// AIGCN_AS_A_69458211111611
// MI455X (gfx1250) — matrix operand order repaired and checked by reading, not compiled, not run on hardware
//
#include <hip/hip_runtime.h>

#define BATCH 2048
#define NPTS  512
#define CFN   30
#define NCH   32      // 31 channels padded to 32
#define HIDN  512
#define KCAT  1536
#define FCN   128

typedef __attribute__((ext_vector_type(16))) _Float16 v16h;
typedef __attribute__((ext_vector_type(8)))  _Float16 v8h;
typedef __attribute__((ext_vector_type(8)))  float    v8f;

static __device__ inline v8f wmma16(v16h a, v16h b, v8f c) {
  return __builtin_amdgcn_wmma_f32_16x16x32_f16(false, a, false, b, (short)0, c, false, false);
}

// A fragment (16x32 f16) from row-major f16 matrix in LDS, leading dim lda.
// Per ISA 7.12.2 the per-lane data is two contiguous 16-byte chunks -> 2x ds_load_b128.
static __device__ inline v16h frag_a_lds(const _Float16* A, int lda, int m0, int k0, int lane) {
  const _Float16* row = A + (m0 + (lane & 15)) * lda + k0 + ((lane >> 4) << 4);
  v8h c0 = *(const v8h*)(row);        // K = kb .. kb+7
  v8h c1 = *(const v8h*)(row + 8);    // K = kb+8 .. kb+15
  v16h a;
#pragma unroll
  for (int e = 0; e < 8; ++e) { a[e] = c0[e]; a[8 + e] = c1[e]; }
  return a;
}

// FAST B fragment (32x16) from f16 weights stored N-major: Wt[n][k], leading dim ldk.
// Per-lane: 16 consecutive K halves = 32 contiguous bytes -> 2x global_load_b128.
static __device__ inline v16h frag_b_h(const _Float16* Wt, int ldk, int k0, int n0, int lane) {
  const _Float16* row = Wt + (size_t)(n0 + (lane & 15)) * ldk + k0 + ((lane >> 4) << 4);
  v8h c0 = *(const v8h*)(row);
  v8h c1 = *(const v8h*)(row + 8);
  v16h b;
#pragma unroll
  for (int e = 0; e < 8; ++e) { b[e] = c0[e]; b[8 + e] = c1[e]; }
  return b;
}

// SLOW B fragment from f32 global stored K-major: W[k][n], leading dim ldn (column gather).
static __device__ inline v16h frag_b_kn(const float* W, int ldn, int k0, int n0, int lane) {
  v16h b;
  const float* col = W + (size_t)(k0 + ((lane >> 4) << 4)) * ldn + (n0 + (lane & 15));
#pragma unroll
  for (int e = 0; e < 16; ++e)
    b[e] = (_Float16)col[(size_t)e * ldn];
  return b;
}

// SLOW B fragment from f32 global stored as W[n][k], leading dim ldk.
static __device__ inline v16h frag_b_nk(const float* W, int ldk, int k0, int n0, int lane) {
  v16h b;
  const float* row = W + (size_t)(n0 + (lane & 15)) * ldk + k0 + ((lane >> 4) << 4);
#pragma unroll
  for (int e = 0; e < 16; ++e)
    b[e] = (_Float16)row[e];
  return b;
}

// One-time: convert/transpose all GEMM weights to f16, N-major (K contiguous).
__global__ __launch_bounds__(256)
void prep_weights(const float* __restrict__ ll1_w, const float* __restrict__ ll2_w,
                  const float* __restrict__ agg_w, const float* __restrict__ dec1_w,
                  const float* __restrict__ mlp_w,
                  _Float16* __restrict__ ll1T, _Float16* __restrict__ ll2T,
                  _Float16* __restrict__ aggT, _Float16* __restrict__ dec1T,
                  _Float16* __restrict__ mlpH) {
  const int total = 3 * 262144 + 65536 + 786432;
  for (int i = blockIdx.x * 256 + threadIdx.x; i < total; i += gridDim.x * 256) {
    if (i < 3 * 262144) {
      int w = i / 262144, r = i - w * 262144;
      int n = r >> 9, k = r & 511;                       // T[n][k] = W[k][n]
      const float* W = (w == 0) ? ll1_w : (w == 1) ? ll2_w : agg_w;
      _Float16*    T = (w == 0) ? ll1T  : (w == 1) ? ll2T  : aggT;
      T[r] = (_Float16)W[k * 512 + n];
    } else if (i < 3 * 262144 + 65536) {
      int r = i - 3 * 262144;
      int n = r >> 9, k = r & 511;                       // dec1T[n][k] = dec1_w[k][n]
      dec1T[r] = (_Float16)dec1_w[k * 128 + n];
    } else {
      int r = i - (3 * 262144 + 65536);                  // mlp_w already N-major
      mlpH[r] = (_Float16)mlp_w[r];
    }
  }
}

template <bool FAST>
__global__ __launch_bounds__(256)
void aigcn_fused(const float* __restrict__ sensor, const float* __restrict__ occp,
                 const float* __restrict__ ll1_w,  const float* __restrict__ ll1_b,
                 const float* __restrict__ ll2_w,  const float* __restrict__ ll2_b,
                 const float* __restrict__ g1_w,   const float* __restrict__ g1_b,
                 const float* __restrict__ g2_w,   const float* __restrict__ g2_b,
                 const float* __restrict__ mlp_w,  const float* __restrict__ mlp_b,
                 const float* __restrict__ agg_w,  const float* __restrict__ agg_b,
                 const float* __restrict__ dec1_w, const float* __restrict__ dec1_b,
                 const float* __restrict__ dec2_w, const float* __restrict__ dec2_b,
                 const float* __restrict__ cd1_w,  const float* __restrict__ cd1_b,
                 const float* __restrict__ cd2_w,  const float* __restrict__ cd2_b,
                 const _Float16* __restrict__ ll1T, const _Float16* __restrict__ ll2T,
                 const _Float16* __restrict__ aggT, const _Float16* __restrict__ dec1T,
                 const _Float16* __restrict__ mlpH,
                 float* __restrict__ out) {
  const int b    = blockIdx.x;
  const int tid  = threadIdx.x;
  const int wave = tid >> 5;
  const int lane = tid & 31;

  // 136 KB LDS -> 2 workgroups resident per 320 KB WGP.
  extern __shared__ char smem[];
  _Float16* sXA  = (_Float16*)(smem);            // 32x512 f16 (x_a)             [0,32768)
  _Float16* sXO  = (_Float16*)(smem);            //   reused after hops: XO f16
  float*    sOcc = (float*)(smem + 32768);       // 512 f32                      [32768,34816)
  float*    sM1  = (float*)(smem + 34816);       // 32 f32
  float*    sM2  = (float*)(smem + 34944);       // 32 f32
  float*    sAdp = (float*)(smem + 35072);       // 32x32 f32                    [35072,39168)
  float*    sdv  = (float*)(smem + 39168);       // 32 f32 (dec2 out)
  float*    stv  = (float*)(smem + 39296);       // 32 f32 (cd1 out)
  // Big region [40960,139264): sE1+sE2 (E/adp) -> sHc (hops/mlp) -> sY + sD1
  _Float16* sE1  = (_Float16*)(smem + 40960);    // 32x512 f16
  _Float16* sE2  = (_Float16*)(smem + 73728);    // 32x512 f16
  _Float16* sHc  = (_Float16*)(smem + 40960);    // 32x1536 f16
  _Float16* sY   = (_Float16*)(smem + 40960);    // 32x512 f16 (after Hcat dead)
  float*    sD1  = (float*)(smem + 73728);       // 32x128 f32

  // ---- Phase 0: build x_a (31x512, padded to 32) + occ row --------------------
  for (int idx = tid; idx < NCH * NPTS; idx += 256) {
    int c = idx >> 9, j = idx & 511;
    float v;
    if (c < CFN)       v = sensor[((size_t)b * NPTS + j) * CFN + c];
    else if (c == CFN) { v = 1.0f - __expf(-occp[(size_t)b * NPTS + j]); sOcc[j] = v; }
    else               v = 0.0f;
    sXA[idx] = (_Float16)v;
  }
  __syncthreads();

  // ---- Phase 1: gates m[c] = relu(tanh(x_a[c]·g_w + g_b)); wave per 4 rows ----
#pragma unroll
  for (int i = 0; i < 4; ++i) {
    int c = wave * 4 + i;
    float a1 = 0.0f, a2 = 0.0f;
    if (c < 31) {
      const _Float16* row = sXA + c * NPTS;
      for (int j = lane; j < NPTS; j += 32) {
        float xv = (float)row[j];
        a1 = fmaf(xv, g1_w[j], a1);
        a2 = fmaf(xv, g2_w[j], a2);
      }
    }
#pragma unroll
    for (int m = 16; m >= 1; m >>= 1) {
      a1 += __shfl_xor(a1, m, 32);
      a2 += __shfl_xor(a2, m, 32);
    }
    if (lane == 0) {
      if (c < 31) {
        sM1[c] = fmaxf(tanhf(a1 + g1_b[0]), 0.0f);
        sM2[c] = fmaxf(tanhf(a2 + g2_b[0]), 0.0f);
      } else if (c == 31) { sM1[31] = 0.0f; sM2[31] = 0.0f; }
    }
  }
  __syncthreads();

  const int mi = (wave >> 2) << 4;        // this wave's M tile row (0 or 16)
  const int qb = (wave & 3) << 3;         // this wave's 8-tile N quarter (tile units)
  const int mb = mi + ((lane >> 4) << 3); // D-fragment base row for stores
  const int nl = lane & 15;

  // ---- Phase 2: z = x_a @ ll_w + b (WMMA, f16 in place), softmax(relu(m*z)) ---
  for (int which = 0; which < 2; ++which) {
    const float*    W  = which ? ll2_w : ll1_w;
    const _Float16* Wh = which ? ll2T  : ll1T;
    const float*    bb = which ? ll2_b : ll1_b;
    const float*    mm = which ? sM2 : sM1;
    _Float16*       E  = which ? sE2 : sE1;

    for (int p = 0; p < 2; ++p) {
      int nb = (qb + p * 4) << 4;
      v8f acc[4] = {};
      for (int kk = 0; kk < NPTS; kk += 32) {
        v16h af = frag_a_lds(sXA, NPTS, mi, kk, lane);
        v16h bf[4];
#pragma unroll
        for (int i = 0; i < 4; ++i) {
          if (FAST) bf[i] = frag_b_h(Wh, NPTS, kk, nb + i * 16, lane);
          else      bf[i] = frag_b_kn(W, NPTS, kk, nb + i * 16, lane);
        }
#pragma unroll
        for (int i = 0; i < 4; ++i) acc[i] = wmma16(af, bf[i], acc[i]);
      }
#pragma unroll
      for (int i = 0; i < 4; ++i) {
        int N = nb + i * 16 + nl;
        float bv = bb[N];
#pragma unroll
        for (int r = 0; r < 8; ++r)
          E[(mb + r) * NPTS + N] = (_Float16)(acc[i][r] + bv);
      }
    }
    __syncthreads();

    for (int c = wave; c < 32; c += 8) {          // in-place softmax(relu(m*z))
      float mc = mm[c];
      _Float16* rowE = E + c * NPTS;
      float mx = 0.0f;
      for (int j = lane; j < NPTS; j += 32) {
        float r = fmaxf(mc * (float)rowE[j], 0.0f);
        rowE[j] = (_Float16)r;
        mx = fmaxf(mx, r);
      }
#pragma unroll
      for (int m = 16; m >= 1; m >>= 1) mx = fmaxf(mx, __shfl_xor(mx, m, 32));
      float s = 0.0f;
      for (int j = lane; j < NPTS; j += 32) {
        float e = __expf((float)rowE[j] - mx);
        rowE[j] = (_Float16)e;
        s += e;
      }
#pragma unroll
      for (int m = 16; m >= 1; m >>= 1) s += __shfl_xor(s, m, 32);
      float inv = 1.0f / s;
      for (int j = lane; j < NPTS; j += 32)
        rowE[j] = (_Float16)((float)rowE[j] * inv);
    }
    __syncthreads();
  }

  // ---- Phase 3: adp = softmax(relu((occ+E1)(31x512) @ (occ+E2)^T(512x31))) ----
  if (wave < 4) {
    int ami = (wave >> 1) << 4, ani = (wave & 1) << 4;
    int M  = ami + nl;
    int Nv = ani + nl;
    v8f acc = {};
    for (int kk = 0; kk < NPTS; kk += 32) {
      v16h af, bf;
      int kbA = kk + ((lane >> 4) << 4);
#pragma unroll
      for (int e = 0; e < 16; ++e) {
        int K = kbA + e;
        af[e] = (_Float16)(sOcc[K] + (float)sE1[M * NPTS + K]);
      }
      int kbB = kk + ((lane >> 4) << 4);
#pragma unroll
      for (int e = 0; e < 16; ++e) {
        int K = kbB + e;
        bf[e] = (_Float16)(sOcc[K] + (float)sE2[Nv * NPTS + K]);
      }
      acc = wmma16(af, bf, acc);
    }
    int amb = ami + ((lane >> 4) << 3);
#pragma unroll
    for (int r = 0; r < 8; ++r) sAdp[(amb + r) * 32 + Nv] = acc[r];
  }
  __syncthreads();

  if (tid < 32) {
    if (tid < 31) {
      float* rowA = sAdp + tid * 32;
      float mx = 0.0f;
      for (int v = 0; v < 31; ++v) { float r = fmaxf(rowA[v], 0.0f); rowA[v] = r; mx = fmaxf(mx, r); }
      float s = 0.0f;
      for (int v = 0; v < 31; ++v) { float e = __expf(rowA[v] - mx); rowA[v] = e; s += e; }
      float inv = 1.0f / s;
      for (int v = 0; v < 31; ++v) rowA[v] *= inv;
      rowA[31] = 0.0f;
    } else {
      for (int v = 0; v < 32; ++v) sAdp[31 * 32 + v] = 0.0f;
    }
  }
  __syncthreads();

  // ---- Phase 4: graph hops -> Hcat (32 x 1536) f16 (overlays E1/E2) ----------
  for (int idx = tid; idx < NCH * KCAT; idx += 256) {   // hop0 copy + zero pad
    int n = idx / KCAT, c = idx - n * KCAT;
    sHc[idx] = (c < NPTS) ? sXA[n * NPTS + c] : (_Float16)0.0f;
  }
  __syncthreads();

  for (int idx = tid; idx < NPTS * 31; idx += 256) {    // hop1: xk1 = x @ adp
    int f = idx & 511, v = idx >> 9;
    float acc = 0.0f;
    for (int n = 0; n < 31; ++n)
      acc = fmaf((float)sXA[n * NPTS + f], sAdp[n * 32 + v], acc);
    sHc[v * KCAT + NPTS + f] = (_Float16)acc;
  }
  __syncthreads();
  for (int idx = tid; idx < NPTS * 31; idx += 256) {    // hop2: xk2 = xk1 @ adp
    int f = idx & 511, v = idx >> 9;
    float acc = 0.0f;
    for (int n = 0; n < 31; ++n)                        // read hop1 from Hcat (f16)
      acc = fmaf((float)sHc[n * KCAT + NPTS + f], sAdp[n * 32 + v], acc);
    sHc[v * KCAT + 2 * NPTS + f] = (_Float16)acc;
  }
  __syncthreads();

  // ---- Phase 5: XO = relu(Hcat @ mlp_w^T + mlp_b)  (32x1536)@(1536x512) -------
  // sXO overlays sXA (dead after hop1).
  for (int p = 0; p < 2; ++p) {
    int nb = (qb + p * 4) << 4;
    v8f acc[4] = {};
    for (int kk = 0; kk < KCAT; kk += 32) {
      v16h af = frag_a_lds(sHc, KCAT, mi, kk, lane);
      v16h bf[4];
#pragma unroll
      for (int i = 0; i < 4; ++i) {
        if (FAST) bf[i] = frag_b_h(mlpH, KCAT, kk, nb + i * 16, lane);
        else      bf[i] = frag_b_nk(mlp_w, KCAT, kk, nb + i * 16, lane);
      }
#pragma unroll
      for (int i = 0; i < 4; ++i) acc[i] = wmma16(af, bf[i], acc[i]);
    }
#pragma unroll
    for (int i = 0; i < 4; ++i) {
      int N = nb + i * 16 + nl;
      float bv = mlp_b[N];
#pragma unroll
      for (int r = 0; r < 8; ++r)
        sXO[(mb + r) * HIDN + N] = (_Float16)fmaxf(acc[i][r] + bv, 0.0f);
    }
  }
  __syncthreads();

  // ---- Phase 6: Y = XO @ agg_w + agg_b  (32x512)@(512x512); sY overlays Hcat --
  for (int p = 0; p < 2; ++p) {
    int nb = (qb + p * 4) << 4;
    v8f acc[4] = {};
    for (int kk = 0; kk < HIDN; kk += 32) {
      v16h af = frag_a_lds(sXO, HIDN, mi, kk, lane);
      v16h bf[4];
#pragma unroll
      for (int i = 0; i < 4; ++i) {
        if (FAST) bf[i] = frag_b_h(aggT, HIDN, kk, nb + i * 16, lane);
        else      bf[i] = frag_b_kn(agg_w, HIDN, kk, nb + i * 16, lane);
      }
#pragma unroll
      for (int i = 0; i < 4; ++i) acc[i] = wmma16(af, bf[i], acc[i]);
    }
#pragma unroll
    for (int i = 0; i < 4; ++i) {
      int N = nb + i * 16 + nl;
      float bv = agg_b[N];
#pragma unroll
      for (int r = 0; r < 8; ++r)
        sY[(mb + r) * HIDN + N] = (_Float16)(acc[i][r] + bv);
    }
  }
  __syncthreads();

  // ---- Phase 7: D1 = relu(Y @ dec1_w + dec1_b)  (32x512)@(512x128) ------------
  {
    int nb = (wave & 3) << 5;     // 2 tiles (32 cols) per wave quarter
    v8f acc[2] = {};
    for (int kk = 0; kk < HIDN; kk += 32) {
      v16h af = frag_a_lds(sY, HIDN, mi, kk, lane);
      v16h bf[2];
#pragma unroll
      for (int i = 0; i < 2; ++i) {
        if (FAST) bf[i] = frag_b_h(dec1T, HIDN, kk, nb + i * 16, lane);
        else      bf[i] = frag_b_kn(dec1_w, FCN, kk, nb + i * 16, lane);
      }
#pragma unroll
      for (int i = 0; i < 2; ++i) acc[i] = wmma16(af, bf[i], acc[i]);
    }
#pragma unroll
    for (int i = 0; i < 2; ++i) {
      int N = nb + i * 16 + nl;
      float bv = dec1_b[N];
#pragma unroll
      for (int r = 0; r < 8; ++r)
        sD1[(mb + r) * FCN + N] = fmaxf(acc[i][r] + bv, 0.0f);
    }
  }
  __syncthreads();

  // ---- Phase 8: dec2 + cd heads -> scalar ------------------------------------
  if (tid < 31) {
    float acc = dec2_b[0];
    for (int k = 0; k < FCN; ++k) acc = fmaf(sD1[tid * FCN + k], dec2_w[k], acc);
    sdv[tid] = acc;
  }
  __syncthreads();
  if (tid < 32) {
    float t = cd1_b[tid];
    for (int n = 0; n < 31; ++n) t = fmaf(sdv[n], cd1_w[n * 32 + tid], t);
    stv[tid] = fmaxf(t, 0.0f);
  }
  __syncthreads();
  if (tid == 0) {
    float s = cd2_b[0];
    for (int q = 0; q < 32; ++q) s = fmaf(stv[q], cd2_w[q], s);
    out[b] = fabsf(s);
  }
}

extern "C" void kernel_launch(void* const* d_in, const int* in_sizes, int n_in,
                              void* d_out, int out_size, void* d_ws, size_t ws_size,
                              hipStream_t stream) {
  (void)in_sizes; (void)n_in; (void)out_size;
  const float* sensor = (const float*)d_in[0];
  const float* occp   = (const float*)d_in[1];
  const float* ll1_w  = (const float*)d_in[2];
  const float* ll1_b  = (const float*)d_in[3];
  const float* ll2_w  = (const float*)d_in[4];
  const float* ll2_b  = (const float*)d_in[5];
  const float* g1_w   = (const float*)d_in[6];
  const float* g1_b   = (const float*)d_in[7];
  const float* g2_w   = (const float*)d_in[8];
  const float* g2_b   = (const float*)d_in[9];
  const float* mlp_w  = (const float*)d_in[10];
  const float* mlp_b  = (const float*)d_in[11];
  const float* agg_w  = (const float*)d_in[12];
  const float* agg_b  = (const float*)d_in[13];
  const float* dec1_w = (const float*)d_in[14];
  const float* dec1_b = (const float*)d_in[15];
  const float* dec2_w = (const float*)d_in[16];
  const float* dec2_b = (const float*)d_in[17];
  const float* cd1_w  = (const float*)d_in[18];
  const float* cd1_b  = (const float*)d_in[19];
  const float* cd2_w  = (const float*)d_in[20];
  const float* cd2_b  = (const float*)d_in[21];

  constexpr size_t SMEM_BYTES = 139264;  // 136 KB LDS -> 2 blocks per WGP
  // f16 weight cache layout in d_ws (halves): ll1T, ll2T, aggT (512x512 each),
  // dec1T (128x512), mlpH (512x1536)
  constexpr size_t WS_HALVES = 3 * 262144 + 65536 + 786432;   // 1,638,400
  constexpr size_t WS_NEEDED = WS_HALVES * sizeof(_Float16);  // ~3.3 MB

  if (ws_size >= WS_NEEDED) {
    _Float16* ws    = (_Float16*)d_ws;
    _Float16* ll1T  = ws;
    _Float16* ll2T  = ws + 262144;
    _Float16* aggT  = ws + 524288;
    _Float16* dec1T = ws + 786432;
    _Float16* mlpH  = ws + 851968;
    prep_weights<<<dim3(6400), dim3(256), 0, stream>>>(
        ll1_w, ll2_w, agg_w, dec1_w, mlp_w, ll1T, ll2T, aggT, dec1T, mlpH);
    aigcn_fused<true><<<dim3(BATCH), dim3(256), SMEM_BYTES, stream>>>(
        sensor, occp, ll1_w, ll1_b, ll2_w, ll2_b, g1_w, g1_b, g2_w, g2_b,
        mlp_w, mlp_b, agg_w, agg_b, dec1_w, dec1_b, dec2_w, dec2_b,
        cd1_w, cd1_b, cd2_w, cd2_b, ll1T, ll2T, aggT, dec1T, mlpH, (float*)d_out);
  } else {
    aigcn_fused<false><<<dim3(BATCH), dim3(256), SMEM_BYTES, stream>>>(
        sensor, occp, ll1_w, ll1_b, ll2_w, ll2_b, g1_w, g1_b, g2_w, g2_b,
        mlp_w, mlp_b, agg_w, agg_b, dec1_w, dec1_b, dec2_w, dec2_b,
        cd1_w, cd1_b, cd2_w, cd2_b, nullptr, nullptr, nullptr, nullptr, nullptr,
        (float*)d_out);
  }
}
